// SelfAttention_86947317941193
// MI455X (gfx1250) — matrix operand order repaired and checked by reading, not compiled, not run on hardware
//
#include <hip/hip_runtime.h>
#include <hip/hip_bf16.h>
#include <math.h>

// Self-attention for MI455X (gfx1250, wave32, WMMA 16x16x32 f16).
// B=4, S=2048, E=1024, H=16, HDIM=64.  ~90 GFLOP total, >> HBM cost (~6us at
// 23.3 TB/s for the 134MB of fp32 I/O) -> matrix-core bound; everything runs
// through v_wmma_f32_16x16x32_f16 (f16 in, f32 accumulate).
//
// Attention is computed *transposed* (S^T = K*Q^T, O^T = V^T*P^T) so the query
// index lands in the lane dimension of the C-fragment layout:
//   - softmax reductions = one shfl_xor(16)
//   - P^T tile -> LDS is one ds_store_b128 per lane, read back as a B-fragment
//   - O^T stores are contiguous 16B per lane
// K/V tiles are staged into double-buffered LDS with the CDNA5 async path
// (global_load_async_to_lds_b128 + s_wait_asynccnt), shared by the 8 waves of
// a block (all on the same (b,h)), overlapping fills with WMMA.

#define EMBED 1024
#define HEADS 16
#define HDIM  64
#define BATCH 4
#define SEQ   2048

#define KPAD 72   // LDS row stride (halves) for K tile (32 x 64), 144B = 16B-aligned
#define VPAD 40   // LDS row stride for V^T tile (64 x 32), 80B
#define PPAD 40   // LDS row stride for per-wave P^T tile (16 x 32), 80B

typedef __attribute__((ext_vector_type(16))) _Float16 v16h;
typedef __attribute__((ext_vector_type(8)))  _Float16 v8h;
typedef __attribute__((ext_vector_type(8)))  float    v8f;
typedef __attribute__((ext_vector_type(4)))  float    v4f;

union V16U { v16h v; v8h h[2]; };

__device__ __forceinline__ v8f zero8() {
  v8f z = {0.f,0.f,0.f,0.f,0.f,0.f,0.f,0.f};
  return z;
}

__device__ __forceinline__ v8f wmma16(v16h a, v16h b, v8f c) {
  // D = A(16x32 f16) * B(32x16 f16) + C(16x16 f32)
  return __builtin_amdgcn_wmma_f32_16x16x32_f16(false, a, false, b,
                                                (short)0, c, false, false);
}

// ---- CDNA5 async global->LDS (ASYNCcnt-tracked), per-lane 16B -------------
__device__ __forceinline__ void async_ld16(unsigned ldsoff, const _Float16* g) {
  asm volatile("global_load_async_to_lds_b128 %0, %1, off"
               :: "v"(ldsoff), "v"(g) : "memory");
}
__device__ __forceinline__ void wait_async0() {
  asm volatile("s_wait_asynccnt 0x0" ::: "memory");
}

// ---- fragment loaders ------------------------------------------------------
// A fragment (16x32 f16): per-lane row pointer; halves 0..15 <- K=2*koff..+15,
// koff = 8*(lane>>4).
__device__ __forceinline__ v16h ldA16(const _Float16* rowp, int koff) {
  V16U u;
  u.h[0] = *(const v8h*)(rowp + 2 * koff);
  u.h[1] = *(const v8h*)(rowp + 2 * koff + 8);
  return u.v;
}
// B fragment (32x16): per-lane 16 contiguous K halves starting at p.
__device__ __forceinline__ v16h ldB16(const _Float16* p) {
  V16U u;
  u.h[0] = *(const v8h*)(p);
  u.h[1] = *(const v8h*)(p + 8);
  return u.v;
}
__device__ __forceinline__ v8h cvt8(const float* p) {
  v4f a = *(const v4f*)(p);
  v4f b = *(const v4f*)(p + 4);
  v8h o;
#pragma unroll
  for (int i = 0; i < 4; ++i) { o[i] = (_Float16)a[i]; o[i + 4] = (_Float16)b[i]; }
  return o;
}
__device__ __forceinline__ v16h ldA32(const float* rowp, int koff) {
  V16U u;
  u.h[0] = cvt8(rowp + 2 * koff);
  u.h[1] = cvt8(rowp + 2 * koff + 8);
  return u.v;
}
__device__ __forceinline__ v16h ldB32(const float* p) {
  V16U u;
  u.h[0] = cvt8(p);
  u.h[1] = cvt8(p + 8);
  return u.v;
}

// ---- weight conversion -----------------------------------------------------
__global__ __launch_bounds__(256) void f32_to_f16_kernel(const float* __restrict__ src,
                                                         _Float16* __restrict__ dst, int n) {
  int i = blockIdx.x * blockDim.x + threadIdx.x;
  if (i < n) dst[i] = (_Float16)src[i];
}

// ---- Q/K projection: Out(b,h,s,d) = X @ W^T (row-major out) ---------------
// Orientation D^T = W * X^T -> C layout (M=e, N=s): e contiguous in r.
__global__ __launch_bounds__(256) void proj_rm_kernel(const float* __restrict__ X,
                                                      const _Float16* __restrict__ Wh,
                                                      _Float16* __restrict__ Out) {
  const int lane = threadIdx.x & 31;
  const int n = lane & 15, hi = lane >> 4;
  const int koff = hi * 8;
  int wid = blockIdx.x * 8 + (threadIdx.x >> 5);
  int st = wid & 127, h = (wid >> 7) & 15, b = wid >> 11;
  int s0 = st * 16;

  const float* xrow = X + (size_t)(b * SEQ + s0 + n) * EMBED + h * HDIM;

  v16h A[4][2];
#pragma unroll
  for (int et = 0; et < 4; ++et)
#pragma unroll
    for (int kf = 0; kf < 2; ++kf)
      A[et][kf] = ldA16(Wh + (et * 16 + n) * HDIM + kf * 32, koff);

  v16h Bf[2];
#pragma unroll
  for (int kf = 0; kf < 2; ++kf)
    Bf[kf] = ldB32(xrow + kf * 32 + hi * 16);

  _Float16* orow = Out + ((size_t)(b * HEADS + h) * SEQ + s0 + n) * HDIM;
#pragma unroll
  for (int et = 0; et < 4; ++et) {
    v8f c = zero8();
    c = wmma16(A[et][0], Bf[0], c);
    c = wmma16(A[et][1], Bf[1], c);
    v8h o;
#pragma unroll
    for (int r = 0; r < 8; ++r) o[r] = (_Float16)c[r];
    *(v8h*)(orow + et * 16 + hi * 8) = o;   // d = et*16 + 8*hi + r
  }
}

// ---- V projection into transposed layout Vt(b,h,d,s) -----------------------
// Orientation D = X * W^T -> C layout (M=s, N=e): s contiguous in r.
__global__ __launch_bounds__(256) void proj_tr_kernel(const float* __restrict__ X,
                                                      const _Float16* __restrict__ Wh,
                                                      _Float16* __restrict__ Vt) {
  const int lane = threadIdx.x & 31;
  const int n = lane & 15, hi = lane >> 4;
  const int koff = hi * 8;
  int wid = blockIdx.x * 8 + (threadIdx.x >> 5);
  int st = wid & 127, h = (wid >> 7) & 15, b = wid >> 11;
  int s0 = st * 16;

  const float* xrow = X + (size_t)(b * SEQ + s0 + n) * EMBED + h * HDIM;

  v16h A[2];
#pragma unroll
  for (int kf = 0; kf < 2; ++kf)
    A[kf] = ldA32(xrow + kf * 32, koff);

  _Float16* vb = Vt + ((size_t)(b * HEADS + h) * HDIM) * SEQ;
#pragma unroll
  for (int et = 0; et < 4; ++et) {
    v8f c = zero8();
#pragma unroll
    for (int kf = 0; kf < 2; ++kf) {
      v16h Bf = ldB16(Wh + (et * 16 + n) * HDIM + kf * 32 + hi * 16);
      c = wmma16(A[kf], Bf, c);
    }
    v8h o;
#pragma unroll
    for (int r = 0; r < 8; ++r) o[r] = (_Float16)c[r];
    *(v8h*)(vb + (size_t)(et * 16 + n) * SEQ + s0 + hi * 8) = o;
  }
}

// ---- flash attention: async double-buffered K/V staging in LDS -------------
// 8 waves/block, all on the same (b,h), consecutive 16-query tiles.
__global__ __launch_bounds__(256) void attn_kernel(const _Float16* __restrict__ Qh,
                                                   const _Float16* __restrict__ Kh,
                                                   const _Float16* __restrict__ Vt,
                                                   _Float16* __restrict__ Attn) {
  __shared__ _Float16 sK[2][32 * KPAD];  // (k-local, d) padded rows
  __shared__ _Float16 sV[2][64 * VPAD];  // (d, k-local) padded rows
  __shared__ _Float16 sP[8][16 * PPAD];  // per-wave P^T (q, k-local)

  const int lane = threadIdx.x & 31;
  const int wv = threadIdx.x >> 5;
  const int n = lane & 15, hi = lane >> 4;
  const int koff = hi * 8;
  _Float16* P = sP[wv];

  int wid = blockIdx.x * 8 + wv;
  int qt = wid & 127, h = (wid >> 7) & 15, b = wid >> 11;
  int q0 = qt * 16;
  size_t bh = (size_t)(b * HEADS + h);
  const _Float16* Qp = Qh + bh * SEQ * HDIM;
  const _Float16* Kp = Kh + bh * SEQ * HDIM;
  const _Float16* Vp = Vt + bh * (size_t)HDIM * SEQ;

  // per-thread async-copy slices: K tile = 32 rows x 8 chunks, V = 64 x 4.
  const int t = threadIdx.x;
  const int kr = t >> 3, kc = t & 7;
  const int vr = t >> 2, vc = t & 3;
  unsigned skoff0 = (unsigned)(size_t)&sK[0][kr * KPAD + kc * 8];
  unsigned skoff1 = (unsigned)(size_t)&sK[1][kr * KPAD + kc * 8];
  unsigned svoff0 = (unsigned)(size_t)&sV[0][vr * VPAD + vc * 8];
  unsigned svoff1 = (unsigned)(size_t)&sV[1][vr * VPAD + vc * 8];
  const _Float16* kg = Kp + (size_t)kr * HDIM + kc * 8;  // + kb*HDIM per step
  const _Float16* vg = Vp + (size_t)vr * SEQ + vc * 8;   // + kb per step

  // Q^T B-fragments (lane -> query column q0+n), resident for all keys.
  v16h BQ[2];
#pragma unroll
  for (int kf = 0; kf < 2; ++kf)
    BQ[kf] = ldB16(Qp + (q0 + n) * HDIM + kf * 32 + hi * 16);

  v8f O[4];
#pragma unroll
  for (int dt = 0; dt < 4; ++dt) O[dt] = zero8();
  float m2 = -3.0e38f;                 // running max (log2 domain)
  float l = 0.f;                       // running sum
  const float SC = 0.0450842245f;      // log2(e) / sqrt(1024)

  // prefill buffer 0
  async_ld16(skoff0, kg);
  async_ld16(svoff0, vg);

  for (int kb = 0; kb < SEQ; kb += 32) {
    int cur = (kb >> 5) & 1;
    wait_async0();          // this wave's fills done
    __syncthreads();        // all waves' fills visible; everyone done with cur^1
    if (kb + 32 < SEQ) {    // refill the other buffer asynchronously
      async_ld16(cur ? skoff0 : skoff1, kg + (size_t)(kb + 32) * HDIM);
      async_ld16(cur ? svoff0 : svoff1, vg + (kb + 32));
    }
    const _Float16* sKc = sK[cur];
    const _Float16* sVc = sV[cur];

    // S^T tiles: M = key (two 16-key tiles), N = query
    v8f S[2];
#pragma unroll
    for (int kt = 0; kt < 2; ++kt) {
      v8f c = zero8();
#pragma unroll
      for (int kf = 0; kf < 2; ++kf) {
        v16h AK = ldA16(sKc + (kt * 16 + n) * KPAD + kf * 32, koff);
        c = wmma16(AK, BQ[kf], c);
      }
      S[kt] = c;
    }
    // online softmax over keys (per query = per lane)
    float mt = m2;
#pragma unroll
    for (int kt = 0; kt < 2; ++kt)
#pragma unroll
      for (int r = 0; r < 8; ++r) {
        float ts = S[kt][r] * SC;
        S[kt][r] = ts;
        mt = fmaxf(mt, ts);
      }
    mt = fmaxf(mt, __shfl_xor(mt, 16, 32));
    float corr = exp2f(m2 - mt);
    m2 = mt;
    float rsum = 0.f;
#pragma unroll
    for (int kt = 0; kt < 2; ++kt) {
      v8h p;
#pragma unroll
      for (int r = 0; r < 8; ++r) {
        float e = exp2f(S[kt][r] - m2);
        rsum += e;
        p[r] = (_Float16)e;
      }
      *(v8h*)(P + n * PPAD + kt * 16 + hi * 8) = p;  // one 16B LDS store
    }
    rsum += __shfl_xor(rsum, 16, 32);
    l = l * corr + rsum;

    v16h BP = ldB16(P + n * PPAD + hi * 16);  // P^T back as B-fragment
#pragma unroll
    for (int dt = 0; dt < 4; ++dt) {
#pragma unroll
      for (int r = 0; r < 8; ++r) O[dt][r] *= corr;
      v16h AV = ldA16(sVc + (dt * 16 + n) * VPAD, koff);
      O[dt] = wmma16(AV, BP, O[dt]);
    }
  }

  float inv = 1.f / l;
  _Float16* orow = Attn + (size_t)(b * SEQ + q0 + n) * EMBED + h * HDIM;
#pragma unroll
  for (int dt = 0; dt < 4; ++dt) {
    v8h o;
#pragma unroll
    for (int r = 0; r < 8; ++r) o[r] = (_Float16)(O[dt][r] * inv);
    *(v8h*)(orow + dt * 16 + hi * 8) = o;   // d = dt*16 + 8*hi + r
  }
}

// ---- output GEMM: Out(m,e) = Attn(m,:) @ Wo^T + bo, fp32 out ---------------
// Orientation D^T = Wo * Attn^T : C layout (M=e, N=m); wave tile 64e x 32m.
// Waves in a block share the Wo stream (same e-macro) -> WGP$ dedup.
__global__ __launch_bounds__(256) void ogemm_kernel(const _Float16* __restrict__ Attn,
                                                    const _Float16* __restrict__ Woh,
                                                    const float* __restrict__ bo,
                                                    float* __restrict__ Out) {
  const int lane = threadIdx.x & 31;
  const int n = lane & 15, hi = lane >> 4;
  const int koff = hi * 8;
  int wid = blockIdx.x * 8 + (threadIdx.x >> 5);
  int mt = wid & 255, et = wid >> 8;      // block: same et, 8 consecutive mt
  int e0 = et * 64, m0 = mt * 32;

  v8f C[2][4];
#pragma unroll
  for (int mi = 0; mi < 2; ++mi)
#pragma unroll
    for (int e4 = 0; e4 < 4; ++e4) C[mi][e4] = zero8();

  for (int kk = 0; kk < EMBED; kk += 32) {
    v16h Bf[2];
#pragma unroll
    for (int mi = 0; mi < 2; ++mi)
      Bf[mi] = ldB16(Attn + (size_t)(m0 + mi * 16 + n) * EMBED + kk + hi * 16);
#pragma unroll
    for (int e4 = 0; e4 < 4; ++e4) {
      v16h Af = ldA16(Woh + (size_t)(e0 + e4 * 16 + n) * EMBED + kk, koff);
#pragma unroll
      for (int mi = 0; mi < 2; ++mi)
        C[mi][e4] = wmma16(Af, Bf[mi], C[mi][e4]);
    }
  }

#pragma unroll
  for (int mi = 0; mi < 2; ++mi)
#pragma unroll
    for (int e4 = 0; e4 < 4; ++e4) {
      int eb = e0 + e4 * 16 + hi * 8;   // 8 consecutive e per lane
      v4f b0 = *(const v4f*)(bo + eb);
      v4f b1 = *(const v4f*)(bo + eb + 4);
      v4f o0, o1;
#pragma unroll
      for (int r = 0; r < 4; ++r) {
        o0[r] = C[mi][e4][r] + b0[r];
        o1[r] = C[mi][e4][r + 4] + b1[r];
      }
      float* op = Out + (size_t)(m0 + mi * 16 + n) * EMBED + eb;
      *(v4f*)op = o0;
      *(v4f*)(op + 4) = o1;
    }
}

// ---- host-side launcher ----------------------------------------------------
extern "C" void kernel_launch(void* const* d_in, const int* in_sizes, int n_in,
                              void* d_out, int out_size, void* d_ws, size_t ws_size,
                              hipStream_t stream) {
  (void)in_sizes; (void)n_in; (void)out_size; (void)ws_size;
  const float* Vx = (const float*)d_in[0];
  const float* Kx = (const float*)d_in[1];
  const float* Qx = (const float*)d_in[2];
  /* d_in[3] = mask, unused (as in reference) */
  const float* Wv = (const float*)d_in[4];
  const float* Wk = (const float*)d_in[5];
  const float* Wq = (const float*)d_in[6];
  const float* Wo = (const float*)d_in[7];
  const float* bo = (const float*)d_in[8];
  float* Out = (float*)d_out;

  char* ws = (char*)d_ws;
  size_t off = 0;
  auto alloc = [&](size_t bytes) -> void* {
    void* p = ws + off;
    off += (bytes + 255) & ~(size_t)255;
    return p;
  };
  const size_t BHSD = (size_t)BATCH * HEADS * SEQ * HDIM;   // 8.4M elems
  _Float16* Wqh = (_Float16*)alloc(HDIM * HDIM * 2);
  _Float16* Wkh = (_Float16*)alloc(HDIM * HDIM * 2);
  _Float16* Wvh = (_Float16*)alloc(HDIM * HDIM * 2);
  _Float16* Woh = (_Float16*)alloc((size_t)EMBED * EMBED * 2);
  _Float16* Qh  = (_Float16*)alloc(BHSD * 2);
  _Float16* Kh  = (_Float16*)alloc(BHSD * 2);
  _Float16* Vt  = (_Float16*)alloc(BHSD * 2);
  _Float16* At  = (_Float16*)alloc((size_t)BATCH * SEQ * EMBED * 2);

  f32_to_f16_kernel<<<(HDIM * HDIM + 255) / 256, 256, 0, stream>>>(Wq, Wqh, HDIM * HDIM);
  f32_to_f16_kernel<<<(HDIM * HDIM + 255) / 256, 256, 0, stream>>>(Wk, Wkh, HDIM * HDIM);
  f32_to_f16_kernel<<<(HDIM * HDIM + 255) / 256, 256, 0, stream>>>(Wv, Wvh, HDIM * HDIM);
  f32_to_f16_kernel<<<(EMBED * EMBED + 255) / 256, 256, 0, stream>>>(Wo, Woh, EMBED * EMBED);

  proj_rm_kernel<<<1024, 256, 0, stream>>>(Qx, Wqh, Qh);
  proj_rm_kernel<<<1024, 256, 0, stream>>>(Kx, Wkh, Kh);
  proj_tr_kernel<<<1024, 256, 0, stream>>>(Vx, Wvh, Vt);

  attn_kernel<<<1024, 256, 0, stream>>>(Qh, Kh, Vt, At);

  ogemm_kernel<<<512, 256, 0, stream>>>(At, Woh, bo, Out);
}
